// AttentionDecoder_37503654429424
// MI455X (gfx1250) — matrix operand order repaired and checked by reading, not compiled, not run on hardware
//
#include <hip/hip_runtime.h>
#include <hip/hip_bf16.h>

// Problem dims (compile-time constants from the reference)
#define T_ 64
#define B_ 32
#define L_ 1024
#define E_ 128
#define V_ 512
#define A_ 512
#define H_ 512

typedef __bf16 bf16_t;
typedef __attribute__((ext_vector_type(2)))  __bf16 v2bf;
typedef __attribute__((ext_vector_type(16))) __bf16 v16bf;
typedef __attribute__((ext_vector_type(8)))  float  v8f;
typedef __attribute__((ext_vector_type(4)))  unsigned int u32x4;
typedef __attribute__((ext_vector_type(4)))  int i32x4;
typedef __attribute__((ext_vector_type(8)))  int i32x8;

__device__ __forceinline__ float fsigmoid(float x) { return 1.0f / (1.0f + __expf(-x)); }
__device__ __forceinline__ float ftanh(float x) {
  float e = __expf(2.0f * x);
  return 1.0f - 2.0f / (e + 1.0f);   // robust for large |x|
}

// ---------------------------------------------------------------------------
// TDM: issue a 2D tensor_load_to_lds (D# per ISA 08_async_tensor §8.3/8.4).
// data_size_code: 0=1B,1=2B,2=4B.  Dims/strides in element units.
// ---------------------------------------------------------------------------
__device__ __forceinline__ void tdm_load_2d(unsigned int lds_addr, const void* gptr,
                                            unsigned int ds_code, unsigned int tile_d0,
                                            unsigned int tile_d1, unsigned int tensor_d0,
                                            unsigned int tensor_d1, unsigned int stride0) {
  unsigned long long ga = (unsigned long long)(uintptr_t)gptr;
  u32x4 g0;
  g0[0] = 1u;                                            // count=1, user mode
  g0[1] = lds_addr;                                      // LDS byte address
  g0[2] = (unsigned int)ga;                              // global_addr[31:0]
  g0[3] = ((unsigned int)(ga >> 32) & 0x01FFFFFFu) | (2u << 30);  // addr[56:32] | type=2
  i32x8 g1;
  g1[0] = (int)(ds_code << 16);                          // wg_mask=0, data_size
  g1[1] = (int)((tensor_d0 & 0xFFFFu) << 16);            // tensor_dim0[15:0]
  g1[2] = (int)((tensor_d0 >> 16) | ((tensor_d1 & 0xFFFFu) << 16));
  g1[3] = (int)((tensor_d1 >> 16) | (tile_d0 << 16));    // tile_dim0
  g1[4] = (int)(tile_d1 & 0xFFFFu);                      // tile_dim1, tile_dim2=0
  g1[5] = (int)stride0;                                  // tensor_dim0_stride[31:0]
  g1[6] = 0; g1[7] = 0;
  i32x4 z4 = {0, 0, 0, 0};
#if defined(__clang_major__) && __clang_major__ >= 23
  i32x8 z8 = {0, 0, 0, 0, 0, 0, 0, 0};
  __builtin_amdgcn_tensor_load_to_lds(g0, g1, z4, z4, z8, 0);
#else
  __builtin_amdgcn_tensor_load_to_lds(g0, g1, z4, z4, 0);
#endif
}

// ---------------------------------------------------------------------------
// Wave-level 16x16 tile GEMM helpers, bf16 WMMA with fp32 accumulate.
// A tile 16x32 bf16: lane<16 -> row M=lane, K {0..7,16..23}; lane>=16 -> K {8..15,24..31}
// B tile 32x16 bf16: lane%16 = col N, lane/16 selects K half
// C/D 16x16 f32:     VGPR r: lanes0-15 -> M=r, lanes16-31 -> M=r+8
// ---------------------------------------------------------------------------
__device__ __forceinline__ v16bf pack_a(float4 a0, float4 a1, float4 a2, float4 a3) {
  v16bf av;
  av[0]=(bf16_t)a0.x;  av[1]=(bf16_t)a0.y;  av[2]=(bf16_t)a0.z;  av[3]=(bf16_t)a0.w;
  av[4]=(bf16_t)a1.x;  av[5]=(bf16_t)a1.y;  av[6]=(bf16_t)a1.z;  av[7]=(bf16_t)a1.w;
  av[8]=(bf16_t)a2.x;  av[9]=(bf16_t)a2.y;  av[10]=(bf16_t)a2.z; av[11]=(bf16_t)a2.w;
  av[12]=(bf16_t)a3.x; av[13]=(bf16_t)a3.y; av[14]=(bf16_t)a3.z; av[15]=(bf16_t)a3.w;
  return av;
}

// Generic K-loop for the small per-step GEMMs: float4 A loads, bf16 B loads.
template <class FA4, class FB>
__device__ __forceinline__ v8f wmma_k_loop(int K, int mrow, int abase, int kbB, int ncol,
                                           FA4 loadA4, FB loadB) {
  v8f acc = {};
  for (int k0 = 0; k0 < K; k0 += 32) {
    v16bf av = pack_a(loadA4(mrow, k0 + abase),      loadA4(mrow, k0 + abase + 4),
                      loadA4(mrow, k0 + 16 + abase), loadA4(mrow, k0 + 16 + abase + 4));
    v16bf bv;
#pragma unroll
    for (int j = 0; j < 16; ++j) bv[j] = loadB(k0 + abase + (j & 7) + ((j >> 3) << 4), ncol);
    acc = __builtin_amdgcn_wmma_f32_16x16x32_bf16(false, av, false, bv,
                                                  (short)0, acc, false, false);
  }
  return acc;
}

__device__ __forceinline__ void store_tile_f32(float* __restrict__ C, int ldc, int m0, int ncol,
                                               const float* __restrict__ bias, v8f acc) {
  int lane = threadIdx.x & 31;
  int mb = (lane >> 4) << 3;
  float bv = bias ? bias[ncol] : 0.0f;
#pragma unroll
  for (int r = 0; r < 8; ++r) C[(size_t)(m0 + mb + r) * ldc + ncol] = acc[r] + bv;
}

__device__ __forceinline__ void store_tile_bf16(bf16_t* __restrict__ C, int ldc, int m0, int ncol,
                                                const float* __restrict__ bias, v8f acc) {
  int lane = threadIdx.x & 31;
  int mb = (lane >> 4) << 3;
  float bv = bias ? bias[ncol] : 0.0f;
#pragma unroll
  for (int r = 0; r < 8; ++r) C[(size_t)(m0 + mb + r) * ldc + ncol] = (bf16_t)(acc[r] + bv);
}

#define TILE_COORDS()                          \
  int lane  = threadIdx.x & 31;                \
  int mrow  = m0 + (lane & 15);                \
  int abase = (lane >> 4) << 3;                \
  int ncol  = n0 + (lane & 15);                \
  int kbB   = (lane >> 4) << 4;

// fp32 -> bf16 weight conversion (one-time)
__global__ void kConvBf16(const float* __restrict__ src, bf16_t* __restrict__ dst, int n) {
  int id = blockIdx.x * blockDim.x + threadIdx.x;
  if (id < n) dst[id] = (bf16_t)src[id];
}

// ---------------------------------------------------------------------------
// enc_feat = encoder_states @ W_h + b_h  -> bf16   (M=32768, N=512, K=512)
// TDM double-buffered pipeline: per 32-deep K slice, wave0 DMAs the
// 128x32 fp32 A panel (16KB) and 32x64 bf16 B panel (4KB) into LDS,
// 8 waves x 4 tiles compute from LDS only.
// grid (256, 8), block 256 (8 waves).
// ---------------------------------------------------------------------------
__global__ void kEncFeat(const float* __restrict__ enc, const bf16_t* __restrict__ Whbf,
                         const float* __restrict__ bh, bf16_t* __restrict__ encf) {
  __shared__ __align__(16) float  sA[2][128 * 32];
  __shared__ __align__(16) bf16_t sB[2][32 * 64];

  const int w    = threadIdx.x >> 5;
  const int lane = threadIdx.x & 31;
  const int m0panel = blockIdx.x * 128;
  const int n0base  = blockIdx.y * 64;
  const int m0 = m0panel + w * 16;
  const int abase = (lane >> 4) << 3;
  const int kbB   = (lane >> 4) << 4;

  unsigned int ldsA0 = (unsigned int)(uintptr_t)&sA[0][0];
  unsigned int ldsA1 = (unsigned int)(uintptr_t)&sA[1][0];
  unsigned int ldsB0 = (unsigned int)(uintptr_t)&sB[0][0];
  unsigned int ldsB1 = (unsigned int)(uintptr_t)&sB[1][0];

  v8f acc[4] = {};

  const int NK = V_ / 32;  // 16 K slices
  if (threadIdx.x < 32) {  // wave 0 drives the TDM
    tdm_load_2d(ldsA0, enc + (size_t)m0panel * V_, 2, 32, 128, V_, B_ * L_, V_);
    tdm_load_2d(ldsB0, Whbf + n0base, 1, 64, 32, A_, V_, A_);
  }
  int buf = 0;
  for (int it = 0; it < NK; ++it) {
    const int k0 = it * 32;
    if (threadIdx.x < 32) {
      if (it + 1 < NK) {
        unsigned int la = buf ? ldsA0 : ldsA1;
        unsigned int lb = buf ? ldsB0 : ldsB1;
        tdm_load_2d(la, enc + (size_t)m0panel * V_ + (k0 + 32), 2, 32, 128, V_, B_ * L_, V_);
        tdm_load_2d(lb, Whbf + (size_t)(k0 + 32) * A_ + n0base, 1, 64, 32, A_, V_, A_);
        __builtin_amdgcn_s_wait_tensorcnt(2);   // current slice's pair done
      } else {
        __builtin_amdgcn_s_wait_tensorcnt(0);
      }
    }
    __syncthreads();

    const float* As = &sA[buf][0] + ((w << 4) + (lane & 15)) * 32;
    v16bf av = pack_a(*(const float4*)(As + abase),      *(const float4*)(As + abase + 4),
                      *(const float4*)(As + abase + 16), *(const float4*)(As + abase + 20));
#pragma unroll
    for (int tt = 0; tt < 4; ++tt) {
      v16bf bv;
#pragma unroll
      for (int j = 0; j < 16; ++j)
        bv[j] = sB[buf][(abase + (j & 7) + ((j >> 3) << 4)) * 64 + tt * 16 + (lane & 15)];
      acc[tt] = __builtin_amdgcn_wmma_f32_16x16x32_bf16(false, av, false, bv,
                                                        (short)0, acc[tt], false, false);
    }
    __syncthreads();
    buf ^= 1;
  }
#pragma unroll
  for (int tt = 0; tt < 4; ++tt)
    store_tile_bf16(encf, A_, m0, n0base + tt * 16 + (lane & 15), bh, acc[tt]);
}

// x = [dec_in_t, ctx] @ W_x + b_x   (M=32, N=128, K=640)
__global__ void kGemmX(const float* __restrict__ din, const float* __restrict__ ctx,
                       const bf16_t* __restrict__ Wxbf, const float* __restrict__ bx,
                       float* __restrict__ x) {
  int wid = threadIdx.x >> 5;             // 16 waves, 1 block
  int m0 = (wid >> 3) * 16, n0 = (wid & 7) * 16;
  TILE_COORDS();
  auto loadA4 = [=](int m, int k) {
    return (k < E_) ? *(const float4*)(din + (size_t)m * E_ + k)
                    : *(const float4*)(ctx + (size_t)m * V_ + (k - E_));
  };
  auto loadB = [=](int k, int n) { return Wxbf[(size_t)k * E_ + n]; };
  v8f acc = wmma_k_loop(E_ + V_, mrow, abase, kbB, ncol, loadA4, loadB);
  store_tile_f32(x, E_, m0, ncol, bx, acc);
}

// gates = x @ W_lstm_k + h @ W_lstm_r + b_lstm  (M=32, N=2048, K=640 fused concat)
__global__ void kGemmGates(const float* __restrict__ x, const float* __restrict__ h,
                           const bf16_t* __restrict__ Wkbf, const bf16_t* __restrict__ Wrbf,
                           const float* __restrict__ bl, float* __restrict__ gates) {
  int wid = blockIdx.x * (blockDim.x >> 5) + (threadIdx.x >> 5);
  const int NT = (4 * H_) / 16;           // 128
  if (wid >= 2 * NT) return;
  int m0 = (wid / NT) * 16, n0 = (wid % NT) * 16;
  TILE_COORDS();
  auto loadA4 = [=](int m, int k) {
    return (k < E_) ? *(const float4*)(x + (size_t)m * E_ + k)
                    : *(const float4*)(h + (size_t)m * H_ + (k - E_));
  };
  auto loadB = [=](int k, int n) {
    return (k < E_) ? Wkbf[(size_t)k * 4 * H_ + n] : Wrbf[(size_t)(k - E_) * 4 * H_ + n];
  };
  v8f acc = wmma_k_loop(E_ + H_, mrow, abase, kbB, ncol, loadA4, loadB);
  store_tile_f32(gates, 4 * H_, m0, ncol, bl, acc);
}

// LSTM elementwise update; also zeroes ctx for this step's attention (H_ == V_)
__global__ void kLstm(const float* __restrict__ gates, float* __restrict__ c,
                      float* __restrict__ h, float* __restrict__ ctx) {
  int id = blockIdx.x * blockDim.x + threadIdx.x;
  if (id >= B_ * H_) return;
  int b = id / H_, j = id % H_;
  const float* g = gates + (size_t)b * 4 * H_;
  float ig = fsigmoid(g[j]);
  float fg = fsigmoid(g[H_ + j]);
  float gg = ftanh(g[2 * H_ + j]);
  float og = fsigmoid(g[3 * H_ + j]);
  float cn = fg * c[id] + ig * gg;
  c[id] = cn;
  h[id] = og * ftanh(cn);
  ctx[id] = 0.0f;                          // B_*V_ == B_*H_
}

// dec_feat = [c, h] @ W_s + b_s  (M=32, N=512, K=1024)
__global__ void kDecFeat(const float* __restrict__ c, const float* __restrict__ h,
                         const bf16_t* __restrict__ Wsbf, const float* __restrict__ bs,
                         float* __restrict__ df) {
  int wid = blockIdx.x * (blockDim.x >> 5) + (threadIdx.x >> 5);
  const int NT = A_ / 16;                  // 32
  if (wid >= 2 * NT) return;
  int m0 = (wid / NT) * 16, n0 = (wid % NT) * 16;
  TILE_COORDS();
  auto loadA4 = [=](int m, int k) {
    return (k < H_) ? *(const float4*)(c + (size_t)m * H_ + k)
                    : *(const float4*)(h + (size_t)m * H_ + (k - H_));
  };
  auto loadB = [=](int k, int n) { return Wsbf[(size_t)k * A_ + n]; };
  v8f acc = wmma_k_loop(2 * H_, mrow, abase, kbB, ncol, loadA4, loadB);
  store_tile_f32(df, A_, m0, ncol, bs, acc);
}

// e[b,l] = sum_a v[a]*tanh(enc_feat + dec_feat + cov*w_cov + b_cov); one wave per (b,l)
__global__ void kEnergy(const bf16_t* __restrict__ encf, const float* __restrict__ df,
                        const float* __restrict__ cov, const float* __restrict__ vvec,
                        const float* __restrict__ wcov, const float* __restrict__ bcov,
                        float* __restrict__ eng) {
  int w = blockIdx.x * (blockDim.x >> 5) + (threadIdx.x >> 5);   // == b*L_ + l
  if (w >= B_ * L_) return;
  int lane = threadIdx.x & 31;
  int b = w / L_;
  const v2bf*   row2 = (const v2bf*)(encf + (size_t)w * A_);
  const float2* d2   = (const float2*)(df + (size_t)b * A_);
  const float2* wc2  = (const float2*)wcov;
  const float2* bc2  = (const float2*)bcov;
  const float2* vv2  = (const float2*)vvec;
  float cv = cov[w];
  float s = 0.0f;
  for (int a2 = lane; a2 < A_ / 2; a2 += 32) {
    v2bf   pk = row2[a2];
    float2 dd = d2[a2], wc = wc2[a2], bc = bc2[a2], vv = vv2[a2];
    float v0 = (float)pk[0] + dd.x + cv * wc.x + bc.x;
    float v1 = (float)pk[1] + dd.y + cv * wc.y + bc.y;
    s += vv.x * ftanh(v0) + vv.y * ftanh(v1);
  }
#pragma unroll
  for (int off = 16; off > 0; off >>= 1) s += __shfl_xor(s, off, 32);
  if (lane == 0) eng[w] = s;
}

// per-batch softmax + mask renorm + coverage update; one 256-thread block per b
__global__ void kSoftmax(const float* __restrict__ eng, const float* __restrict__ mask,
                         float* __restrict__ cov, float* __restrict__ attn,
                         float* __restrict__ attn_out) {
  __shared__ float sd[256];
  int b = blockIdx.x, tid = threadIdx.x;
  float ev[4], ex[4], p[4];
  float mx = -1e30f;
#pragma unroll
  for (int j = 0; j < 4; ++j) { ev[j] = eng[(size_t)b * L_ + tid + j * 256]; mx = fmaxf(mx, ev[j]); }
  sd[tid] = mx; __syncthreads();
  for (int s = 128; s > 0; s >>= 1) { if (tid < s) sd[tid] = fmaxf(sd[tid], sd[tid + s]); __syncthreads(); }
  mx = sd[0]; __syncthreads();
  float sum = 0.0f;
#pragma unroll
  for (int j = 0; j < 4; ++j) { ex[j] = __expf(ev[j] - mx); sum += ex[j]; }
  sd[tid] = sum; __syncthreads();
  for (int s = 128; s > 0; s >>= 1) { if (tid < s) sd[tid] += sd[tid + s]; __syncthreads(); }
  sum = sd[0]; __syncthreads();
  float inv1 = 1.0f / sum, s2 = 0.0f;
#pragma unroll
  for (int j = 0; j < 4; ++j) { p[j] = ex[j] * inv1 * mask[(size_t)b * L_ + tid + j * 256]; s2 += p[j]; }
  sd[tid] = s2; __syncthreads();
  for (int s = 128; s > 0; s >>= 1) { if (tid < s) sd[tid] += sd[tid + s]; __syncthreads(); }
  s2 = sd[0]; __syncthreads();
  float inv2 = 1.0f / s2;
#pragma unroll
  for (int j = 0; j < 4; ++j) {
    int l = tid + j * 256;
    float a = p[j] * inv2;
    attn[(size_t)b * L_ + l] = a;
    attn_out[(size_t)b * L_ + l] = a;
    cov[(size_t)b * L_ + l] += a;
  }
}

// ctx[b,v] = sum_l attn[b,l] * enc[b,l,v]; grid (B*8) x 128 threads, float4 cols
__global__ void kCtxAccum(const float* __restrict__ attn, const float* __restrict__ enc,
                          float* __restrict__ ctx) {
  int b = blockIdx.x >> 3, chunk = blockIdx.x & 7;
  int v4 = threadIdx.x;                    // 0..127 -> cols 4*v4..4*v4+3
  float ax = 0.f, ay = 0.f, az = 0.f, aw = 0.f;
  int l0 = chunk * (L_ / 8);
  for (int l = l0; l < l0 + L_ / 8; ++l) {
    float a = attn[(size_t)b * L_ + l];
    float4 ev = *(const float4*)(enc + ((size_t)b * L_ + l) * V_ + 4 * v4);
    ax += a * ev.x; ay += a * ev.y; az += a * ev.z; aw += a * ev.w;
  }
  float* cp = ctx + (size_t)b * V_ + 4 * v4;
  atomicAdd(cp + 0, ax); atomicAdd(cp + 1, ay);
  atomicAdd(cp + 2, az); atomicAdd(cp + 3, aw);
}

// p_gen[b] = sigmoid([ctx,c,h,x] . W_pg + b_pg); one wave per b
__global__ void kPgen(const float* __restrict__ ctx, const float* __restrict__ c,
                      const float* __restrict__ h, const float* __restrict__ x,
                      const float* __restrict__ Wpg, const float* __restrict__ bpg,
                      float* __restrict__ pg_out) {
  int b = threadIdx.x >> 5;                // 1024-thread block: 32 waves
  int lane = threadIdx.x & 31;
  const int F = V_ + 2 * H_ + E_;          // 1664
  float s = 0.0f;
  for (int k = lane; k < F; k += 32) {
    float f;
    if (k < V_)             f = ctx[(size_t)b * V_ + k];
    else if (k < V_ + H_)   f = c[(size_t)b * H_ + (k - V_)];
    else if (k < V_ + 2*H_) f = h[(size_t)b * H_ + (k - V_ - H_)];
    else                    f = x[(size_t)b * E_ + (k - V_ - 2 * H_)];
    s += f * Wpg[k];
  }
#pragma unroll
  for (int off = 16; off > 0; off >>= 1) s += __shfl_xor(s, off, 32);
  if (lane == 0) pg_out[b] = fsigmoid(s + bpg[0]);
}

// out = [h, ctx] @ W_out + b_out  (M=32, N=512, K=1024), written straight into d_out
__global__ void kGemmOut(const float* __restrict__ h, const float* __restrict__ ctx,
                         const bf16_t* __restrict__ Wobf, const float* __restrict__ bo,
                         float* __restrict__ out) {
  int wid = blockIdx.x * (blockDim.x >> 5) + (threadIdx.x >> 5);
  const int NT = H_ / 16;                  // 32
  if (wid >= 2 * NT) return;
  int m0 = (wid / NT) * 16, n0 = (wid % NT) * 16;
  TILE_COORDS();
  auto loadA4 = [=](int m, int k) {
    return (k < H_) ? *(const float4*)(h + (size_t)m * H_ + k)
                    : *(const float4*)(ctx + (size_t)m * V_ + (k - H_));
  };
  auto loadB = [=](int k, int n) { return Wobf[(size_t)k * H_ + n]; };
  v8f acc = wmma_k_loop(H_ + V_, mrow, abase, kbB, ncol, loadA4, loadB);
  store_tile_f32(out, H_, m0, ncol, bo, acc);
}

__global__ void kInit(float* __restrict__ ctx, float* __restrict__ cov) {
  int id = blockIdx.x * blockDim.x + threadIdx.x;
  if (id < B_ * V_) ctx[id] = 0.0f;
  if (id < B_ * L_) cov[id] = 0.0f;
}

__global__ void kFinal(const float* __restrict__ h, const float* __restrict__ c,
                       const float* __restrict__ cov, float* __restrict__ oh,
                       float* __restrict__ oc, float* __restrict__ ocov) {
  int id = blockIdx.x * blockDim.x + threadIdx.x;
  if (id < B_ * H_) { oh[id] = h[id]; oc[id] = c[id]; }
  if (id < B_ * L_) ocov[id] = cov[id];
}

extern "C" void kernel_launch(void* const* d_in, const int* in_sizes, int n_in,
                              void* d_out, int out_size, void* d_ws, size_t ws_size,
                              hipStream_t stream) {
  (void)in_sizes; (void)n_in; (void)out_size; (void)ws_size;
  const float* dec_in  = (const float*)d_in[0];
  const float* init_h  = (const float*)d_in[1];
  const float* init_c  = (const float*)d_in[2];
  const float* enc     = (const float*)d_in[3];
  const float* mask    = (const float*)d_in[4];
  const float* Wh      = (const float*)d_in[5];
  const float* bh      = (const float*)d_in[6];
  const float* vvec    = (const float*)d_in[7];
  const float* wcov    = (const float*)d_in[8];
  const float* bcov    = (const float*)d_in[9];
  const float* Ws      = (const float*)d_in[10];
  const float* bs      = (const float*)d_in[11];
  const float* Wx      = (const float*)d_in[12];
  const float* bx      = (const float*)d_in[13];
  const float* Wk      = (const float*)d_in[14];
  const float* Wr      = (const float*)d_in[15];
  const float* bl      = (const float*)d_in[16];
  const float* Wpg     = (const float*)d_in[17];
  const float* bpg     = (const float*)d_in[18];
  const float* Wo      = (const float*)d_in[19];
  const float* bo      = (const float*)d_in[20];

  // Workspace: bf16 enc_feat (33.5 MB, L2-resident) + bf16 weights (5.4 MB) + fp32 state
  bf16_t* encf = (bf16_t*)d_ws;
  bf16_t* Whbf = encf + (size_t)B_ * L_ * A_;
  bf16_t* Wxbf = Whbf + (size_t)V_ * A_;
  bf16_t* Wkbf = Wxbf + (size_t)(E_ + V_) * E_;
  bf16_t* Wrbf = Wkbf + (size_t)E_ * 4 * H_;
  bf16_t* Wsbf = Wrbf + (size_t)H_ * 4 * H_;
  bf16_t* Wobf = Wsbf + (size_t)2 * H_ * A_;
  float* f = (float*)(Wobf + (size_t)(H_ + V_) * H_);
  float* h     = f; f += B_ * H_;
  float* c     = f; f += B_ * H_;
  float* ctx   = f; f += B_ * V_;
  float* cov   = f; f += B_ * L_;
  float* x     = f; f += B_ * E_;
  float* gates = f; f += B_ * 4 * H_;
  float* df    = f; f += B_ * A_;
  float* eng   = f; f += B_ * L_;
  float* attn  = f; f += B_ * L_;

  // d_out regions (flat, return order): outs, h, c, attns, pgens, cov
  float* out = (float*)d_out;
  float* out_outs  = out;
  float* out_h     = out + (size_t)T_ * B_ * H_;
  float* out_c     = out_h + (size_t)B_ * H_;
  float* out_attns = out_c + (size_t)B_ * H_;
  float* out_pgens = out_attns + (size_t)T_ * B_ * L_;
  float* out_cov   = out_pgens + (size_t)T_ * B_;

  // one-time weight conversions to bf16
  auto conv = [&](const float* s, bf16_t* d, int n) {
    kConvBf16<<<dim3((n + 255) / 256), dim3(256), 0, stream>>>(s, d, n);
  };
  conv(Wh, Whbf, V_ * A_);
  conv(Wx, Wxbf, (E_ + V_) * E_);
  conv(Wk, Wkbf, E_ * 4 * H_);
  conv(Wr, Wrbf, H_ * 4 * H_);
  conv(Ws, Wsbf, 2 * H_ * A_);
  conv(Wo, Wobf, (H_ + V_) * H_);

  hipMemcpyAsync(h, init_h, (size_t)B_ * H_ * sizeof(float), hipMemcpyDeviceToDevice, stream);
  hipMemcpyAsync(c, init_c, (size_t)B_ * H_ * sizeof(float), hipMemcpyDeviceToDevice, stream);
  kInit<<<dim3((B_ * L_ + 255) / 256), dim3(256), 0, stream>>>(ctx, cov);

  // One-time TDM-fed WMMA GEMM: 256 M-panels x 8 N-groups
  kEncFeat<<<dim3(256, 8), dim3(256), 0, stream>>>(enc, Whbf, bh, encf);

  for (int t = 0; t < T_; ++t) {
    kGemmX<<<dim3(1), dim3(512), 0, stream>>>(dec_in + (size_t)t * B_ * E_, ctx, Wxbf, bx, x);
    kGemmGates<<<dim3(16), dim3(512), 0, stream>>>(x, h, Wkbf, Wrbf, bl, gates);
    kLstm<<<dim3(64), dim3(256), 0, stream>>>(gates, c, h, ctx);
    kDecFeat<<<dim3(4), dim3(512), 0, stream>>>(c, h, Wsbf, bs, df);
    kEnergy<<<dim3(4096), dim3(256), 0, stream>>>(encf, df, cov, vvec, wcov, bcov, eng);
    kSoftmax<<<dim3(B_), dim3(256), 0, stream>>>(eng, mask, cov, attn,
                                                 out_attns + (size_t)t * B_ * L_);
    kCtxAccum<<<dim3(B_ * 8), dim3(128), 0, stream>>>(attn, enc, ctx);
    kPgen<<<dim3(1), dim3(1024), 0, stream>>>(ctx, c, h, x, Wpg, bpg, out_pgens + (size_t)t * B_);
    kGemmOut<<<dim3(4), dim3(512), 0, stream>>>(h, ctx, Wobf, bo, out_outs + (size_t)t * B_ * H_);
  }

  kFinal<<<dim3((B_ * L_ + 255) / 256), dim3(256), 0, stream>>>(h, c, cov, out_h, out_c, out_cov);
}
